// LargeMarginModule_CosineLoss_35278861369532
// MI455X (gfx1250) — matrix operand order repaired and checked by reading, not compiled, not run on hardware
//
#include <hip/hip_runtime.h>
#include <hip/hip_bf16.h>

typedef __attribute__((ext_vector_type(16))) _Float16 v16h;
typedef __attribute__((ext_vector_type(8)))  _Float16 v8h;
typedef __attribute__((ext_vector_type(8)))  float    v8f;
typedef __attribute__((ext_vector_type(4)))  int      v4i;

#define B_ROWS 8192
#define DIM    1024
#define NCLS   10000
#define BM 128
#define BN 128
#define BK 32
#define LDK 40               /* padded LDS K-stride (halves): 80B = 16B-aligned, 20-bank stride */
#define NT_N 79              /* ceil(10000/128) */
#define NT_M 64              /* 8192/128 */
#define NPAD (NT_N * BN)     /* 10112 */
#define KT_STEPS (DIM / BK)  /* 32 */
#define MSCALED 16.0f        /* 0.5 * 32 */

#if __has_builtin(__builtin_amdgcn_global_load_async_to_lds_b128)
#define HAVE_ASYNC 1
#else
#define HAVE_ASYNC 0
#endif

__device__ __forceinline__ void cp16(const _Float16* g, _Float16* l) {
#if HAVE_ASYNC
  __builtin_amdgcn_global_load_async_to_lds_b128(
      (__attribute__((address_space(1))) v4i*)(g),
      (__attribute__((address_space(3))) v4i*)(l), 0, 0);
#else
  *(v8h*)l = *(const v8h*)g;
#endif
}

__device__ __forceinline__ void wait_async_le(int prevInFlight) {
#if HAVE_ASYNC
  if (prevInFlight) asm volatile("s_wait_asynccnt 0x4" ::: "memory");
  else              asm volatile("s_wait_asynccnt 0x0" ::: "memory");
#else
  (void)prevInFlight;
#endif
}

// ---------------- Kernel 1: out = 32 * x / max(||x||, eps), cast to f16 ----------------
__global__ __launch_bounds__(256) void k_rownorm(const float* __restrict__ x,
                                                 _Float16* __restrict__ A16) {
  const int row = blockIdx.x;
  const int tid = threadIdx.x;
  const float* xr = x + (size_t)row * DIM;
  float v[4];
  float ss = 0.f;
#pragma unroll
  for (int j = 0; j < 4; ++j) {
    v[j] = xr[tid + 256 * j];
    ss += v[j] * v[j];
  }
  // wave32 reduction
#pragma unroll
  for (int off = 16; off > 0; off >>= 1) ss += __shfl_xor(ss, off, 32);
  __shared__ float red[8];
  __shared__ float s_scale;
  if ((tid & 31) == 0) red[tid >> 5] = ss;
  __syncthreads();
  if (tid == 0) {
    float t = 0.f;
#pragma unroll
    for (int i = 0; i < 8; ++i) t += red[i];
    s_scale = 32.0f / fmaxf(sqrtf(t), 1e-12f);
  }
  __syncthreads();
  const float sc = s_scale;
  _Float16* ar = A16 + (size_t)row * DIM;
#pragma unroll
  for (int j = 0; j < 4; ++j) ar[tid + 256 * j] = (_Float16)(v[j] * sc);
}

// ---------------- Kernel 2: W fp32 [C,D] -> f16 [NPAD,D], zero-padded rows ----------------
__global__ __launch_bounds__(256) void k_convw(const float* __restrict__ W,
                                               _Float16* __restrict__ W16) {
  const size_t g = ((size_t)blockIdx.x * 256 + threadIdx.x) * 8;
  const int row = (int)(g / DIM);
  if (row < NCLS) {
#pragma unroll
    for (int j = 0; j < 8; ++j) W16[g + j] = (_Float16)W[g + j];
  } else {
#pragma unroll
    for (int j = 0; j < 8; ++j) W16[g + j] = (_Float16)0.f;
  }
}

// ---------------- Kernel 3: WMMA GEMM + fused margin/bias epilogue ----------------
union Frag { v16h v; v8h h[2]; };

__global__ __launch_bounds__(256) void k_gemm(const _Float16* __restrict__ A16,
                                              const _Float16* __restrict__ W16,
                                              const float* __restrict__ bvec,
                                              const int* __restrict__ targets,
                                              const float* __restrict__ Wf32,
                                              float* __restrict__ out) {
  __shared__ _Float16 sA[2][BM * LDK];
  __shared__ _Float16 sB[2][BN * LDK];

  const int tid  = threadIdx.x;
  const int wave = tid >> 5;
  const int lane = tid & 31;
  const int l16  = lane & 15;
  const int halfSel = lane >> 4;          // 0: lanes 0-15, 1: lanes 16-31
  const int wM = wave & 1;                // 2 waves along M (64 rows each)
  const int wN = wave >> 1;               // 4 waves along N (32 cols each)
  const int mBase = blockIdx.y * BM;
  const int nBase = blockIdx.x * BN;

  v8f acc[4][2] = {};                     // 4 m-tiles x 2 n-tiles of 16x16 f32

  // Stage one 128x32 A tile + 128x32 B tile: 512 chunks of 16B each per matrix,
  // 2 chunks per thread per matrix -> 4 async ops/thread/tile.
  auto issue = [&](int buf, int kt) {
#pragma unroll
    for (int i = 0; i < 2; ++i) {
      const int c   = tid + 256 * i;      // 0..511
      const int row = c >> 2;
      const int kc  = c & 3;
      cp16(A16 + (size_t)(mBase + row) * DIM + kt * BK + kc * 8,
           &sA[buf][row * LDK + kc * 8]);
      cp16(W16 + (size_t)(nBase + row) * DIM + kt * BK + kc * 8,
           &sB[buf][row * LDK + kc * 8]);
    }
  };

  issue(0, 0);

  for (int kt = 0; kt < KT_STEPS; ++kt) {
    const int cur = kt & 1;
    const bool more = (kt + 1 < KT_STEPS);
    if (more) issue(cur ^ 1, kt + 1);     // prefetch next tile into other buffer
    wait_async_le(more ? 1 : 0);          // my tile-kt copies complete
    __syncthreads();                      // everyone's tile-kt copies complete

    // Per-lane fragment gathers per ISA layout tables (CDNA5 ISA §7.12.2)
    Frag a[4], bfr[2];
    const int ka = halfSel * 8;           // A: hi lanes hold K+8 / K+24
    const int kb = halfSel * 8;           // B: hi lanes hold K+8 / K+24
#pragma unroll
    for (int mi = 0; mi < 4; ++mi) {
      const _Float16* p = &sA[cur][(wM * 64 + mi * 16 + l16) * LDK];
      a[mi].h[0] = *(const v8h*)(p + ka);
      a[mi].h[1] = *(const v8h*)(p + 16 + ka);
    }
#pragma unroll
    for (int ni = 0; ni < 2; ++ni) {
      const _Float16* p = &sB[cur][(wN * 32 + ni * 16 + l16) * LDK + kb];
      bfr[ni].h[0] = *(const v8h*)(p);
      bfr[ni].h[1] = *(const v8h*)(p + 16);
    }
#pragma unroll
    for (int mi = 0; mi < 4; ++mi)
#pragma unroll
      for (int ni = 0; ni < 2; ++ni)
        acc[mi][ni] = __builtin_amdgcn_wmma_f32_16x16x32_f16(
            false, a[mi].v, false, bfr[ni].v, (short)0, acc[mi][ni], false, false);

    __syncthreads();                      // done reading buf[cur] before it is refilled
  }

  // Epilogue: pred = acc + b[n]; loss = pred - 16 * W[n, targets[m]]
#pragma unroll
  for (int mi = 0; mi < 4; ++mi) {
    const int mRow = mBase + wM * 64 + mi * 16 + halfSel * 8;  // VGPR r <-> M = mRow + r
    int tIdx[8];
#pragma unroll
    for (int r = 0; r < 8; ++r) tIdx[r] = targets[mRow + r];
#pragma unroll
    for (int ni = 0; ni < 2; ++ni) {
      const int n = nBase + wN * 32 + ni * 16 + l16;
      if (n < NCLS) {
        const float bn = bvec[n];
        const float* wcol = Wf32 + (size_t)n * DIM;
#pragma unroll
        for (int r = 0; r < 8; ++r) {
          const float pred = acc[mi][ni][r] + bn;
          const size_t o = (size_t)(mRow + r) * NCLS + n;
          out[o] = pred - MSCALED * wcol[tIdx[r]];           // out_for_loss
          out[(size_t)B_ROWS * NCLS + o] = pred;             // out_for_predict
        }
      }
    }
  }
}

extern "C" void kernel_launch(void* const* d_in, const int* in_sizes, int n_in,
                              void* d_out, int out_size, void* d_ws, size_t ws_size,
                              hipStream_t stream) {
  const float* x   = (const float*)d_in[0];
  const int*   tg  = (const int*)d_in[1];   // targets (int32 per harness convention)
  const float* W   = (const float*)d_in[2];
  const float* b   = (const float*)d_in[3];
  float* out = (float*)d_out;               // [loss (B*C), predict (B*C)]

  // Workspace: A16 (16 MB) + W16 padded (20.7 MB)
  _Float16* A16 = (_Float16*)d_ws;
  _Float16* W16 = (_Float16*)((char*)d_ws + (size_t)B_ROWS * DIM * sizeof(_Float16));

  k_rownorm<<<B_ROWS, 256, 0, stream>>>(x, A16);
  k_convw<<<(NPAD * DIM) / 2048, 256, 0, stream>>>(W, W16);
  k_gemm<<<dim3(NT_N, NT_M), 256, 0, stream>>>(A16, W16, b, tg, W, out);
}
